// Qwen2MoeDecoderLayer_74457553043887
// MI455X (gfx1250) — matrix operand order repaired and checked by reading, not compiled, not run on hardware
//
#include <hip/hip_runtime.h>
#include <hip/hip_bf16.h>
#include <stdint.h>

// ---------------------------------------------------------------------------
// CDNA5 (gfx1250) WMMA types & helpers
// ---------------------------------------------------------------------------
typedef __attribute__((ext_vector_type(16))) __bf16 v16bf;
typedef __attribute__((ext_vector_type(8)))  float  v8f;

union AFrag { uint32_t u[8]; uint4 q[2]; v16bf v; };  // 16x32 A / 32x16 B bf16 fragment
union CFrag { float    f[8]; v8f   v; };              // 16x16 f32 C/D fragment

// RNE scalar convert (memory-bound elementwise kernels only)
__device__ __forceinline__ uint16_t f32_bf16(float x) {
  uint32_t u = __float_as_uint(x);
  return (uint16_t)((u + 0x7FFFu + ((u >> 16) & 1u)) >> 16);
}
// One-instruction packed truncation convert: {hi.bf16, lo.bf16} via v_perm_b32
__device__ __forceinline__ uint32_t pack2_bf16_fast(float lo, float hi) {
  return __builtin_amdgcn_perm(__float_as_uint(hi), __float_as_uint(lo), 0x07060302u);
}
__device__ __forceinline__ float bf16_f32(uint16_t b) {
  return __uint_as_float(((uint32_t)b) << 16);
}

#define WMMA_BF16(A, B, C) \
  __builtin_amdgcn_wmma_f32_16x16x32_bf16(false, (A), false, (B), (short)0, (C), false, false)

// CDNA5 async copy: 16B per lane, global -> LDS, tracked by ASYNCcnt
__device__ __forceinline__ void async_load_b128_to_lds(uint32_t lds_addr, uint64_t gaddr) {
  asm volatile("global_load_async_to_lds_b128 %0, %1, off"
               :: "v"(lds_addr), "v"(gaddr) : "memory");
}
__device__ __forceinline__ void wait_asynccnt0() {
  asm volatile("s_wait_asynccnt 0x0" ::: "memory");
}

// ---------------------------------------------------------------------------
// Model constants
// ---------------------------------------------------------------------------
static constexpr int  kT    = 2048;   // B*S tokens
static constexpr int  kH    = 2048;   // hidden
static constexpr int  kS    = 1024;   // sequence
static constexpr int  kNH   = 16;     // heads
static constexpr int  kHD   = 128;    // head dim
static constexpr int  kNE   = 8;      // experts
static constexpr int  kMOEI = 1408;
static constexpr int  kSHI  = 5632;

// ---------------------------------------------------------------------------
// RMSNorm -> bf16
// ---------------------------------------------------------------------------
__global__ __launch_bounds__(256)
void rmsnorm_bf16_kernel(const float* __restrict__ x, const float* __restrict__ w,
                         uint16_t* __restrict__ out, int H) {
  const int row = blockIdx.x;
  const float* xr = x + (size_t)row * H;
  float s = 0.f;
  for (int i = threadIdx.x; i < H; i += 256) { float v = xr[i]; s += v * v; }
  __shared__ float red[256];
  red[threadIdx.x] = s; __syncthreads();
  for (int st = 128; st > 0; st >>= 1) {
    if (threadIdx.x < st) red[threadIdx.x] += red[threadIdx.x + st];
    __syncthreads();
  }
  const float inv = rsqrtf(red[0] / (float)H + 1e-6f);
  for (int i = threadIdx.x; i < H; i += 256)
    out[(size_t)row * H + i] = f32_bf16(xr[i] * inv * w[i]);
}

// ---------------------------------------------------------------------------
// General WMMA GEMM:  C[M,N](f32) (+)= op( A[M,K](bf16) @ B[K,N](f32->bf16) )
//   block = 256 threads (8 waves, 4m x 2n), block tile 128x128, wave tile 32x64
//   A tile (128x32 bf16): double-buffered LDS via global_load_async_to_lds_b128
//     (async DMA engine, ASYNCcnt; 2 instructions per wave per k-step)
//   B tile (32x128 f32): double-buffered LDS as packed bf16 k-pairs
//     (global float4 x2 adjacent k rows -> v_perm pack -> ds_store)
//   fragment reads: 2 x ds_read_b128 each; 8 x v_wmma per k-step of 32;
//   one barrier per k-step. FULLN instantiation has no bounds guards.
// ---------------------------------------------------------------------------
#define BS_STRIDE 20   // B LDS row stride in dwords  (80 B: 16B-aligned, bank-spread)
#define AS_STRIDE 40   // A LDS row stride in bf16    (80 B: 16B-aligned, bank-spread)

template <bool FULLN>
__global__ __launch_bounds__(256)
void gemm_bf16_wmma(const uint16_t* __restrict__ A, const float* __restrict__ Bw,
                    const float* __restrict__ bias, const float* __restrict__ resid,
                    const float* __restrict__ rowScale, int rsStride,
                    float* __restrict__ C, int M, int N, int K, int accumulate) {
  const int tid   = threadIdx.x;
  const int lane  = tid & 31;
  const int wave  = tid >> 5;
  const int wm    = wave >> 1;                  // 0..3
  const int wn    = wave & 1;                   // 0..1
  const int nblk0 = blockIdx.x * 128;
  const int mblk0 = blockIdx.y * 128;
  const int mblk  = mblk0 + wm * 32;
  const int nblk  = nblk0 + wn * 64;
  const int l15   = lane & 15;
  const int lhalf = lane >> 4;

  __shared__ __align__(16) uint32_t Bs[2][128 * BS_STRIDE];
  __shared__ __align__(16) uint16_t As[2][128 * AS_STRIDE];

  // ---- B staging coords: 512 (p, n4) units, 2 per thread (p0 and p0+8)
  const int p0 = tid >> 5;                      // k-pair 0..7
  const int n4 = (tid & 31) * 4;                // 4 consecutive columns

  auto stageB = [&](int k0, int buf) {
#pragma unroll
    for (int i = 0; i < 2; ++i) {
      const int p = p0 + i * 8;
      const float* r0 = Bw + (size_t)(k0 + 2 * p) * N + nblk0 + n4;
      float4 x0, x1;
      if (FULLN) {
        x0 = *(const float4*)r0;
        x1 = *(const float4*)(r0 + N);
        __builtin_prefetch(r0 + (size_t)32 * N, 0, 1);   // speculative: fault-free
      } else {
        float t0[4], t1[4];
#pragma unroll
        for (int j = 0; j < 4; ++j) {
          const int n = nblk0 + n4 + j;
          const bool ok = n < N;
          const float* q = Bw + (size_t)(k0 + 2 * p) * N + (ok ? n : 0);
          t0[j] = ok ? q[0] : 0.f;
          t1[j] = ok ? q[N] : 0.f;
        }
        x0 = make_float4(t0[0], t0[1], t0[2], t0[3]);
        x1 = make_float4(t1[0], t1[1], t1[2], t1[3]);
      }
      uint32_t* bb = Bs[buf];
      bb[(n4 + 0) * BS_STRIDE + p] = pack2_bf16_fast(x0.x, x1.x);
      bb[(n4 + 1) * BS_STRIDE + p] = pack2_bf16_fast(x0.y, x1.y);
      bb[(n4 + 2) * BS_STRIDE + p] = pack2_bf16_fast(x0.z, x1.z);
      bb[(n4 + 3) * BS_STRIDE + p] = pack2_bf16_fast(x0.w, x1.w);
    }
  };

  // ---- A staging coords: 16 async b128 instrs per tile, 2 per wave.
  //      instr t covers rows t*8 + lane/4, 16 B at column (lane&3)*8 elements.
  const int ar0 = (wave * 2 + 0) * 8 + (lane >> 2);
  const int ar1 = (wave * 2 + 1) * 8 + (lane >> 2);
  const int ac  = (lane & 3) * 8;                           // bf16 element col
  const uint16_t* aG0 = A + (size_t)(mblk0 + ar0) * K + ac; // + k0 per step
  const uint16_t* aG1 = A + (size_t)(mblk0 + ar1) * K + ac;
  const uint32_t aL0 = (uint32_t)(uintptr_t)&As[0][ar0 * AS_STRIDE + ac];
  const uint32_t aL1 = (uint32_t)(uintptr_t)&As[0][ar1 * AS_STRIDE + ac];
  const uint32_t aBufOff = (uint32_t)(128 * AS_STRIDE * 2); // bytes between buffers

  auto stageA = [&](int k0, int buf) {
    const uint32_t bofs = buf ? aBufOff : 0u;
    async_load_b128_to_lds(aL0 + bofs, (uint64_t)(uintptr_t)(aG0 + k0));
    async_load_b128_to_lds(aL1 + bofs, (uint64_t)(uintptr_t)(aG1 + k0));
  };

  CFrag acc[2][4];
#pragma unroll
  for (int i = 0; i < 2; ++i)
#pragma unroll
    for (int j = 0; j < 4; ++j)
#pragma unroll
      for (int v = 0; v < 8; ++v) acc[i][j].f[v] = 0.f;

  stageA(0, 0);
  stageB(0, 0);
  wait_asynccnt0();
  __syncthreads();

  int cur = 0;
  for (int k0 = 0; k0 < K; k0 += 32) {
    // ---- A fragments from LDS (vgpr pairs contiguous -> ds_read_b128 x2)
    AFrag a[2];
#pragma unroll
    for (int fm = 0; fm < 2; ++fm) {
      const int mrel = wm * 32 + fm * 16 + l15;
      const uint16_t* ap = &As[cur][mrel * AS_STRIDE + lhalf * 16];
      a[fm].q[0] = *(const uint4*)(ap);
      a[fm].q[1] = *(const uint4*)(ap + 8);
    }
    // ---- B fragments from LDS: vgpr v holds k-pair p = lhalf*8 + v (contiguous)
    AFrag b[4];
#pragma unroll
    for (int fn = 0; fn < 4; ++fn) {
      const int nl = wn * 64 + fn * 16 + l15;
      const uint32_t* src = &Bs[cur][nl * BS_STRIDE + lhalf * 8];
      b[fn].q[0] = *(const uint4*)(src);       // ds_read_b128
      b[fn].q[1] = *(const uint4*)(src + 4);   // ds_read_b128
    }
    // ---- stage next tile into the other buffer (overlaps with WMMA below)
    if (k0 + 32 < K) {
      stageA(k0 + 32, cur ^ 1);
      stageB(k0 + 32, cur ^ 1);
    }

#pragma unroll
    for (int fm = 0; fm < 2; ++fm)
#pragma unroll
      for (int fn = 0; fn < 4; ++fn)
        acc[fm][fn].v = WMMA_BF16(a[fm].v, b[fn].v, acc[fm][fn].v);

    wait_asynccnt0();   // own async copies landed in LDS
    __syncthreads();    // all waves' reads of cur done; nxt visible
    cur ^= 1;
  }

  // ---- epilogue (C layout: lane holds col n=l15; rows = vgpr + 8*lhalf)
#pragma unroll
  for (int fm = 0; fm < 2; ++fm)
#pragma unroll
    for (int fn = 0; fn < 4; ++fn) {
      const int n = nblk + fn * 16 + l15;
      if (!FULLN && n >= N) continue;
      const float bv = bias ? bias[n] : 0.f;
#pragma unroll
      for (int v = 0; v < 8; ++v) {
        const int m = mblk + fm * 16 + lhalf * 8 + v;
        if (!FULLN && m >= M) continue;
        float val = acc[fm][fn].f[v] + bv;
        const size_t idx = (size_t)m * N + n;
        if (resid)    val += resid[idx];
        if (rowScale) val *= rowScale[(size_t)m * rsStride];
        if (accumulate) C[idx] += val; else C[idx] = val;
      }
    }
}

// ---------------------------------------------------------------------------
// RoPE on q/k (f32 in, bf16 out), layout (b,s,h*128+d)
// ---------------------------------------------------------------------------
__global__ void rope_qk_kernel(const float* __restrict__ qf, const float* __restrict__ kf,
                               uint16_t* __restrict__ qo, uint16_t* __restrict__ ko) {
  const int idx = blockIdx.x * blockDim.x + threadIdx.x;   // t*1024 + h*64 + i
  if (idx >= kT * kNH * 64) return;
  const int i = idx & 63;
  const int h = (idx >> 6) & 15;
  const int t = idx >> 10;
  const int s = t & (kS - 1);
  const float inv = __powf(1.0e6f, -(float)(2 * i) / 128.0f);
  float sn, cs; __sincosf((float)s * inv, &sn, &cs);
  const size_t base = (size_t)t * kH + h * kHD;
  const float q1 = qf[base + i], q2 = qf[base + 64 + i];
  qo[base + i]      = f32_bf16(q1 * cs - q2 * sn);
  qo[base + 64 + i] = f32_bf16(q2 * cs + q1 * sn);
  const float k1 = kf[base + i], k2 = kf[base + 64 + i];
  ko[base + i]      = f32_bf16(k1 * cs - k2 * sn);
  ko[base + 64 + i] = f32_bf16(k2 * cs + k1 * sn);
}

// V (b,s,h,d) f32 -> Vt (b,h,d,s) bf16   (so PV contraction reads contiguous keys)
__global__ void v_transpose_kernel(const float* __restrict__ vf, uint16_t* __restrict__ vt) {
  const int idx = blockIdx.x * blockDim.x + threadIdx.x;   // ((b*16+h)*128+d)*1024+s
  if (idx >= kT * kH) return;
  const int s = idx & 1023;
  const int d = (idx >> 10) & 127;
  const int h = (idx >> 17) & 15;
  const int b = idx >> 21;
  vt[idx] = f32_bf16(vf[((size_t)(b * kS + s)) * kH + h * kHD + d]);
}

// ---------------------------------------------------------------------------
// Flash attention: one wave per 16-query tile, online softmax, WMMA both GEMMs
//   grid = (B*NH, S/(16*8)), block = 256 (8 waves)
// ---------------------------------------------------------------------------
__global__ __launch_bounds__(256)
void attn_fa_kernel(const uint16_t* __restrict__ qb, const uint16_t* __restrict__ kb,
                    const uint16_t* __restrict__ vt, uint16_t* __restrict__ ctx) {
  const int bh = blockIdx.x, b = bh >> 4, h = bh & 15;
  const int wave = threadIdx.x >> 5, lane = threadIdx.x & 31;
  const int l15 = lane & 15, lhalf = lane >> 4;
  const int qb0 = (blockIdx.y * 8 + wave) * 16;

  __shared__ uint32_t pshare[8][16][16];   // per-wave 16x32 bf16 P tile

  // Q fragments over d=0..127 (4 x 16x32)
  AFrag qfr[4];
  {
    const size_t qrow = ((size_t)(b * kS + qb0 + l15)) * kH + h * kHD;
#pragma unroll
    for (int db = 0; db < 4; ++db) {
      const uint16_t* qp = qb + qrow + db * 32 + lhalf * 8;
      qfr[db].q[0] = *(const uint4*)(qp);
      qfr[db].q[1] = *(const uint4*)(qp + 16);
    }
  }

  CFrag acc[8];
#pragma unroll
  for (int nb = 0; nb < 8; ++nb)
#pragma unroll
    for (int v = 0; v < 8; ++v) acc[nb].f[v] = 0.f;
  float rmax[8], rsum[8];
#pragma unroll
  for (int v = 0; v < 8; ++v) { rmax[v] = -3.0e38f; rsum[v] = 0.f; }

  const float scale = 0.08838834764831845f;  // 1/sqrt(128)

  for (int kb0 = 0; kb0 <= qb0 + 15; kb0 += 32) {
    // ---- scores S = Q @ K^T for 32 keys (two 16-key groups)
    CFrag sf[2];
#pragma unroll
    for (int g = 0; g < 2; ++g) {
#pragma unroll
      for (int v = 0; v < 8; ++v) sf[g].f[v] = 0.f;
      const int key = kb0 + g * 16 + l15;
      const size_t krow = ((size_t)(b * kS + key)) * kH + h * kHD;
#pragma unroll
      for (int db = 0; db < 4; ++db) {
        AFrag kf;
        const uint16_t* kp = kb + krow + db * 32 + lhalf * 8;
        kf.q[0] = *(const uint4*)(kp);
        kf.q[1] = *(const uint4*)(kp + 16);
        sf[g].v = WMMA_BF16(qfr[db].v, kf.v, sf[g].v);
      }
    }
    // ---- scale + causal mask + running row-max
    float nm[8];
#pragma unroll
    for (int v = 0; v < 8; ++v) nm[v] = rmax[v];
#pragma unroll
    for (int g = 0; g < 2; ++g)
#pragma unroll
      for (int v = 0; v < 8; ++v) {
        float s = sf[g].f[v] * scale;
        const int key = kb0 + g * 16 + l15;
        const int qi  = qb0 + lhalf * 8 + v;
        if (key > qi) s = -3.0e38f;
        sf[g].f[v] = s;
        nm[v] = fmaxf(nm[v], s);
      }
#pragma unroll
    for (int d = 1; d < 16; d <<= 1)
#pragma unroll
      for (int v = 0; v < 8; ++v) nm[v] = fmaxf(nm[v], __shfl_xor(nm[v], d, 32));
    // ---- rescale accumulators, exponentiate, row-sum
    float psum[8];
#pragma unroll
    for (int v = 0; v < 8; ++v) {
      const float corr = __expf(rmax[v] - nm[v]);
      rsum[v] *= corr;
      rmax[v] = nm[v];
      psum[v] = 0.f;
#pragma unroll
      for (int nb = 0; nb < 8; ++nb) acc[nb].f[v] *= corr;
    }
#pragma unroll
    for (int g = 0; g < 2; ++g)
#pragma unroll
      for (int v = 0; v < 8; ++v) {
        const float p = __expf(sf[g].f[v] - nm[v]);
        sf[g].f[v] = p;
        psum[v] += p;
      }
#pragma unroll
    for (int d = 1; d < 16; d <<= 1)
#pragma unroll
      for (int v = 0; v < 8; ++v) psum[v] += __shfl_xor(psum[v], d, 32);
#pragma unroll
    for (int v = 0; v < 8; ++v) rsum[v] += psum[v];
    // ---- P through LDS into A-fragment layout (per-wave region, wave-internal dep)
    {
      uint16_t* pw = (uint16_t*)&pshare[wave][0][0];
#pragma unroll
      for (int g = 0; g < 2; ++g)
#pragma unroll
        for (int v = 0; v < 8; ++v)
          pw[(lhalf * 8 + v) * 32 + g * 16 + l15] =
              (uint16_t)(__float_as_uint(sf[g].f[v]) >> 16);   // fast truncate
    }
    AFrag pa;
#pragma unroll
    for (int v = 0; v < 4; ++v) pa.u[v]     = pshare[wave][l15][lhalf * 4 + v];
#pragma unroll
    for (int v = 0; v < 4; ++v) pa.u[4 + v] = pshare[wave][l15][8 + lhalf * 4 + v];
    // ---- acc += P(16x32) @ V(32x128)
#pragma unroll
    for (int nb = 0; nb < 8; ++nb) {
      AFrag vf;
      const int d = nb * 16 + l15;
      const uint16_t* vp = vt + (((size_t)bh * kHD + d) * kS) + kb0 + lhalf * 8;
      vf.q[0] = *(const uint4*)(vp);
      vf.q[1] = *(const uint4*)(vp + 16);
      acc[nb].v = WMMA_BF16(pa.v, vf.v, acc[nb].v);
    }
  }

  // ---- finalize: ctx (b,s,h*128+d) bf16
#pragma unroll
  for (int nb = 0; nb < 8; ++nb) {
    const int d = nb * 16 + l15;
#pragma unroll
    for (int v = 0; v < 8; ++v) {
      const int qi = qb0 + lhalf * 8 + v;
      ctx[((size_t)(b * kS + qi)) * kH + h * kHD + d] = f32_bf16(acc[nb].f[v] / rsum[v]);
    }
  }
}

// ---------------------------------------------------------------------------
// Router: softmax over 8 logits, keep top-2 probs (not renormalized)
// ---------------------------------------------------------------------------
__global__ void router_top2_kernel(const float* __restrict__ logits, float* __restrict__ combine) {
  const int t = blockIdx.x * blockDim.x + threadIdx.x;
  if (t >= kT) return;
  float l[8]; float mx = -3.0e38f;
#pragma unroll
  for (int e = 0; e < 8; ++e) { l[e] = logits[t * 8 + e]; mx = fmaxf(mx, l[e]); }
  float sum = 0.f;
#pragma unroll
  for (int e = 0; e < 8; ++e) { l[e] = __expf(l[e] - mx); sum += l[e]; }
  const float inv = 1.f / sum;
#pragma unroll
  for (int e = 0; e < 8; ++e) l[e] *= inv;
  int i1 = 0;
#pragma unroll
  for (int e = 1; e < 8; ++e) if (l[e] > l[i1]) i1 = e;
  int i2 = -1;
#pragma unroll
  for (int e = 0; e < 8; ++e) if (e != i1 && (i2 < 0 || l[e] > l[i2])) i2 = e;
#pragma unroll
  for (int e = 0; e < 8; ++e) combine[t * 8 + e] = (e == i1 || e == i2) ? l[e] : 0.f;
}

// silu(g) * u -> bf16
__global__ void silu_mul_kernel(const float* __restrict__ g, const float* __restrict__ u,
                                uint16_t* __restrict__ out, int n) {
  const int i = blockIdx.x * blockDim.x + threadIdx.x;
  if (i >= n) return;
  const float gv = g[i];
  out[i] = f32_bf16(gv / (1.f + __expf(-gv)) * u[i]);
}

// per-token sigmoid(h . w_sh_eg)
__global__ __launch_bounds__(256)
void expert_gate_kernel(const uint16_t* __restrict__ h, const float* __restrict__ w,
                        float* __restrict__ eg) {
  const int t = blockIdx.x;
  float s = 0.f;
  for (int i = threadIdx.x; i < kH; i += 256) s += bf16_f32(h[(size_t)t * kH + i]) * w[i];
  __shared__ float red[256];
  red[threadIdx.x] = s; __syncthreads();
  for (int st = 128; st > 0; st >>= 1) {
    if (threadIdx.x < st) red[threadIdx.x] += red[threadIdx.x + st];
    __syncthreads();
  }
  if (threadIdx.x == 0) eg[t] = 1.f / (1.f + __expf(-red[0]));
}

__global__ void add2_kernel(const float* __restrict__ a, const float* __restrict__ b,
                            float* __restrict__ out, int n) {
  const int i = blockIdx.x * blockDim.x + threadIdx.x;
  if (i < n) out[i] = a[i] + b[i];
}

// ---------------------------------------------------------------------------
// Host orchestration
// ---------------------------------------------------------------------------
extern "C" void kernel_launch(void* const* d_in, const int* in_sizes, int n_in,
                              void* d_out, int out_size, void* d_ws, size_t ws_size,
                              hipStream_t stream) {
  const float* x        = (const float*)d_in[0];
  const float* w_ln1    = (const float*)d_in[1];
  const float* wq       = (const float*)d_in[2];
  const float* bq       = (const float*)d_in[3];
  const float* wk       = (const float*)d_in[4];
  const float* bk       = (const float*)d_in[5];
  const float* wv       = (const float*)d_in[6];
  const float* bv       = (const float*)d_in[7];
  const float* wo       = (const float*)d_in[8];
  const float* w_ln2    = (const float*)d_in[9];
  const float* w_router = (const float*)d_in[10];
  const float* w_gate_e = (const float*)d_in[11];
  const float* w_up_e   = (const float*)d_in[12];
  const float* w_down_e = (const float*)d_in[13];
  const float* w_sh_g   = (const float*)d_in[14];
  const float* w_sh_u   = (const float*)d_in[15];
  const float* w_sh_d   = (const float*)d_in[16];
  const float* w_sh_eg  = (const float*)d_in[17];

  // workspace bump allocator
  char* ws = (char*)d_ws;
  size_t off = 0;
  auto alloc = [&](size_t bytes) -> char* {
    char* p = ws + off;
    off += (bytes + 255) & ~(size_t)255;
    return p;
  };
  uint16_t* Hbf    = (uint16_t*)alloc((size_t)kT * kH * 2);
  float*    Qf     = (float*)   alloc((size_t)kT * kH * 4);
  float*    Kf     = (float*)   alloc((size_t)kT * kH * 4);
  float*    Vf     = (float*)   alloc((size_t)kT * kH * 4);
  uint16_t* Qbf    = (uint16_t*)alloc((size_t)kT * kH * 2);
  uint16_t* Kbf    = (uint16_t*)alloc((size_t)kT * kH * 2);
  uint16_t* Vt     = (uint16_t*)alloc((size_t)kT * kH * 2);
  uint16_t* Ctx    = (uint16_t*)alloc((size_t)kT * kH * 2);
  float*    X1     = (float*)   alloc((size_t)kT * kH * 4);
  uint16_t* H2bf   = (uint16_t*)alloc((size_t)kT * kH * 2);
  float*    Logits = (float*)   alloc((size_t)kT * kNE * 4);
  float*    Combine= (float*)   alloc((size_t)kT * kNE * 4);
  float*    Ge     = (float*)   alloc((size_t)kT * kMOEI * 4);
  float*    Ue     = (float*)   alloc((size_t)kT * kMOEI * 4);
  uint16_t* Abf    = (uint16_t*)alloc((size_t)kT * kMOEI * 2);
  float*    Routed = (float*)   alloc((size_t)kT * kH * 4);
  float*    Shg    = (float*)   alloc((size_t)kT * kSHI * 4);
  float*    Shu    = (float*)   alloc((size_t)kT * kSHI * 4);
  uint16_t* Shabf  = (uint16_t*)alloc((size_t)kT * kSHI * 2);
  float*    Eg     = (float*)   alloc((size_t)kT * 4);

  const dim3 blk(256);

  // 1) h = rmsnorm(x, w_ln1)  (bf16)
  rmsnorm_bf16_kernel<<<kT, blk, 0, stream>>>(x, w_ln1, Hbf, kH);

  // 2) QKV projections
  {
    dim3 g(kH / 128, kT / 128);
    gemm_bf16_wmma<true><<<g, blk, 0, stream>>>(Hbf, wq, bq, nullptr, nullptr, 0, Qf, kT, kH, kH, 0);
    gemm_bf16_wmma<true><<<g, blk, 0, stream>>>(Hbf, wk, bk, nullptr, nullptr, 0, Kf, kT, kH, kH, 0);
    gemm_bf16_wmma<true><<<g, blk, 0, stream>>>(Hbf, wv, bv, nullptr, nullptr, 0, Vf, kT, kH, kH, 0);
  }

  // 3) RoPE q,k -> bf16 ; V -> (b,h,d,s) bf16
  rope_qk_kernel<<<(kT * kNH * 64 + 255) / 256, blk, 0, stream>>>(Qf, Kf, Qbf, Kbf);
  v_transpose_kernel<<<(kT * kH + 255) / 256, blk, 0, stream>>>(Vf, Vt);

  // 4) attention
  attn_fa_kernel<<<dim3(2 * kNH, kS / 128), blk, 0, stream>>>(Qbf, Kbf, Vt, Ctx);

  // 5) X1 = x + ctx @ wo
  gemm_bf16_wmma<true><<<dim3(kH / 128, kT / 128), blk, 0, stream>>>(
      Ctx, wo, nullptr, x, nullptr, 0, X1, kT, kH, kH, 0);

  // 6) h2 = rmsnorm(X1, w_ln2)
  rmsnorm_bf16_kernel<<<kT, blk, 0, stream>>>(X1, w_ln2, H2bf, kH);

  // 7) router logits + top-2 combine (N=8 -> guarded instantiation)
  gemm_bf16_wmma<false><<<dim3(1, kT / 128), blk, 0, stream>>>(
      H2bf, w_router, nullptr, nullptr, nullptr, 0, Logits, kT, kNE, kH, 0);
  router_top2_kernel<<<(kT + 255) / 256, blk, 0, stream>>>(Logits, Combine);

  // 8) routed experts (dense, accumulated with per-token combine weight)
  hipMemsetAsync(Routed, 0, (size_t)kT * kH * 4, stream);
  for (int e = 0; e < kNE; ++e) {
    const float* wg = w_gate_e + (size_t)e * kH * kMOEI;
    const float* wu = w_up_e   + (size_t)e * kH * kMOEI;
    const float* wd = w_down_e + (size_t)e * kMOEI * kH;
    dim3 gi(kMOEI / 128, kT / 128);
    gemm_bf16_wmma<true><<<gi, blk, 0, stream>>>(H2bf, wg, nullptr, nullptr, nullptr, 0, Ge, kT, kMOEI, kH, 0);
    gemm_bf16_wmma<true><<<gi, blk, 0, stream>>>(H2bf, wu, nullptr, nullptr, nullptr, 0, Ue, kT, kMOEI, kH, 0);
    silu_mul_kernel<<<(kT * kMOEI + 255) / 256, blk, 0, stream>>>(Ge, Ue, Abf, kT * kMOEI);
    gemm_bf16_wmma<true><<<dim3(kH / 128, kT / 128), blk, 0, stream>>>(
        Abf, wd, nullptr, nullptr, Combine + e, kNE, Routed, kT, kH, kMOEI, 1);
  }

  // 9) shared expert (accumulated into Routed with sigmoid gate)
  {
    dim3 gs(kSHI / 128, kT / 128);
    gemm_bf16_wmma<true><<<gs, blk, 0, stream>>>(H2bf, w_sh_g, nullptr, nullptr, nullptr, 0, Shg, kT, kSHI, kH, 0);
    gemm_bf16_wmma<true><<<gs, blk, 0, stream>>>(H2bf, w_sh_u, nullptr, nullptr, nullptr, 0, Shu, kT, kSHI, kH, 0);
    silu_mul_kernel<<<(kT * kSHI + 255) / 256, blk, 0, stream>>>(Shg, Shu, Shabf, kT * kSHI);
    expert_gate_kernel<<<kT, blk, 0, stream>>>(H2bf, w_sh_eg, Eg);
    gemm_bf16_wmma<true><<<dim3(kH / 128, kT / 128), blk, 0, stream>>>(
        Shabf, w_sh_d, nullptr, nullptr, Eg, 1, Routed, kT, kH, kSHI, 1);
  }

  // 10) out = X1 + Routed
  add2_kernel<<<(kT * kH + 255) / 256, blk, 0, stream>>>(X1, Routed, (float*)d_out, kT * kH);
}
